// ThompsonSamplingGraphTransformerGFN_59476707115262
// MI455X (gfx1250) — matrix operand order repaired and checked by reading, not compiled, not run on hardware
//
#include <hip/hip_runtime.h>
#include <math.h>

typedef __attribute__((ext_vector_type(16))) _Float16 v16h;
typedef __attribute__((ext_vector_type(8)))  float    v8f;

#define DEV __device__ __forceinline__
#define CDIV(a,b) (((a)+(b)-1)/(b))

constexpr int   NND   = 65536;          // nodes
constexpr int   NE    = 131072;         // edges
constexpr int   NG    = 2048;           // graphs
constexpr int   EMB   = 64;
constexpr int   HC    = 128;            // HEADS*EMB
constexpr int   NTOT  = NND + NG;       // 67584 (nodes + virtual nodes)
constexpr int   EVN   = NE + 2 * NND;   // 262144 edges before self loops
constexpr int   EAUG  = EVN + NTOT;     // 329728 augmented edges
constexpr float LEAK    = 0.01f;
constexpr float EPS_GEN = 1e-7f;
constexpr float EPS_LN  = 1e-5f;

// All GEMM row counts are multiples of 128 (one block's row span):
static_assert(NND  % 128 == 0 && NE % 128 == 0 && NG % 128 == 0, "");
static_assert(NTOT % 128 == 0 && EAUG % 128 == 0, "");

// ---------------------------------------------------------------------------
// src/dst of augmented edge list, computed on the fly (never materialized)
// ---------------------------------------------------------------------------
DEV void edge_sd(int e, const int* __restrict__ ei, const int* __restrict__ batch,
                 int& s, int& d) {
  if (e < NE)             { s = ei[e];              d = ei[NE + e]; }
  else if (e < NE + NND)  { int u = e - NE;         s = u;                 d = batch[u] + NND; }
  else if (e < EVN)       { int u = e - NE - NND;   s = batch[u] + NND;    d = u; }
  else                    { int u = e - EVN;        s = u;                 d = u; }
}

DEV void atomicMaxF(float* addr, float val) {
  int* ai = (int*)addr;
  int cur = __float_as_int(*addr);
  while (__int_as_float(cur) < val) {
    int prev = atomicCAS(ai, cur, __float_as_int(val));
    if (prev == cur) break;
    cur = prev;
  }
}

DEV void cvt8(v16h& v, int base, float4 a, float4 b) {
  v[base + 0] = (_Float16)a.x; v[base + 1] = (_Float16)a.y;
  v[base + 2] = (_Float16)a.z; v[base + 3] = (_Float16)a.w;
  v[base + 4] = (_Float16)b.x; v[base + 5] = (_Float16)b.y;
  v[base + 6] = (_Float16)b.z; v[base + 7] = (_Float16)b.w;
}

// ---------------------------------------------------------------------------
// WMMA GEMM:  C[M,Nout] (= or +=) act(A[M,K] @ W[K,Nout] + bias)
//  * K compile-time; all predicates wave-uniform (EXEC stays all-1s for WMMA).
//  * A row stride == K; rows loaded as 4x global_load_b128 per K-step.
//  * B tile (32K x 64N) staged once per block per K-step into LDS as f16,
//    column-major with 80B pitch -> each lane's 32B fragment is two
//    16B-aligned ds_load_b128, shared by all 8 waves (8x less VMEM traffic).
//  * M % 128 == 0 for every call site -> no row guards.
//  * blockDim 256 = 8 waves; wave -> 16 rows x 64 cols (4 f32 16x16 accs),
//    grid.y tiles Nout in chunks of 64.
// ---------------------------------------------------------------------------
template<int K, int ACT, bool ACC>
__global__ __launch_bounds__(256) void gemm_f16(
    const float* __restrict__ A,
    const float* __restrict__ W, int ldw,
    const float* __restrict__ bias,
    float* __restrict__ C, int ldc, int M)
{
  constexpr int BSTR = 40;                       // halfs per column (80 B pitch)
  __shared__ __align__(16) _Float16 Bs[64 * BSTR];

  const int tid  = threadIdx.x;
  const int lane = tid & 31;
  const int wave = tid >> 5;
  const int half = lane >> 4;      // which 16-lane half of the wave
  const int r    = lane & 15;
  const int rowBase = blockIdx.x * 128 + wave * 16;
  const int colBase = blockIdx.y * 64;
  const float* __restrict__ Arow = A + (size_t)(rowBase + r) * K;

  v8f acc[4] = {};
  for (int k0 = 0; k0 < K; k0 += 32) {
    // ---- cooperative stage of B tile to LDS (f16, layout Bs[n][kk]) ----
#pragma unroll
    for (int i = 0; i < 8; ++i) {
      const int sidx = tid + i * 256;
      const int kk = sidx >> 6;                  // 0..31, wave-uniform
      const int n  = sidx & 63;                  // coalesced along N
      float wv = 0.f;
      if (k0 + kk < K) wv = W[(size_t)(k0 + kk) * ldw + (colBase + n)];
      Bs[n * BSTR + kk] = (_Float16)wv;
    }
    __syncthreads();

    // A tile 16x32 f16: elems 0-7 <- K[k0+half*8 ..), elems 8-15 <- K[k0+16+half*8 ..)
    v16h av = {};
    {
      const float4* p0 = (const float4*)(Arow + k0 + half * 8);
      cvt8(av, 0, p0[0], p0[1]);
      if (k0 + 16 < K) {                         // wave-uniform (false only for K=16)
        const float4* p1 = (const float4*)(Arow + k0 + 16 + half * 8);
        cvt8(av, 8, p1[0], p1[1]);
        __builtin_prefetch(Arow + k0 + 32, 0, 0);    // global_prefetch_b8
      }
    }
#pragma unroll
    for (int nt = 0; nt < 4; ++nt) {
      // B fragment: lane(0-15)=col, elem j<8 -> kk = half*8 + j, elem j>=8 -> kk = 16 + half*8 + (j-8)
      const int nl = nt * 16 + r;
      union { uint4 u4[2]; v16h v; } bu;
      bu.u4[0] = *(const uint4*)&Bs[nl * BSTR + half * 8];
      bu.u4[1] = *(const uint4*)&Bs[nl * BSTR + 16 + half * 8];
      acc[nt] = __builtin_amdgcn_wmma_f32_16x16x32_f16(
          false, av, false, bu.v, (short)0, acc[nt], false, false);
    }
    __syncthreads();
  }
  // C layout: elem j -> row = rowBase + j + half*8 ; col = colBase + nt*16 + r
#pragma unroll
  for (int nt = 0; nt < 4; ++nt) {
    const int col = colBase + nt * 16 + r;
    const float bval = bias ? bias[col] : 0.f;
#pragma unroll
    for (int j = 0; j < 8; ++j) {
      const int mm = rowBase + j + half * 8;
      float v = acc[nt][j] + bval;
      if (ACT == 1) v = (v > 0.f) ? v : LEAK * v;
      if (ACC) C[(size_t)mm * ldc + col] += v;
      else     C[(size_t)mm * ldc + col]  = v;
    }
  }
}

template<int K, int ACT, bool ACC>
static inline void launch_gemm(const float* A, const float* W, int ldw, const float* bias,
                               float* C, int ldc, int M, int Nout, hipStream_t s) {
  dim3 grd(M / 128, Nout / 64, 1), blk(256, 1, 1);
  gemm_f16<K, ACT, ACC><<<grd, blk, 0, s>>>(A, W, ldw, bias, C, ldc, M);
}

// ---------------------------------------------------------------------------
// Elementwise / scatter kernels
// ---------------------------------------------------------------------------
__global__ void fill_f32(float* __restrict__ p, float v, int n) {
  int i = blockIdx.x * blockDim.x + threadIdx.x;
  if (i < n) p[i] = v;
}

__global__ void count_nodes(const int* __restrict__ batch,
                            float* __restrict__ cntA, float* __restrict__ cntN) {
  int v = blockIdx.x * blockDim.x + threadIdx.x;
  if (v >= NND) return;
  atomicAdd(&cntA[batch[v]], 1.f);
  atomicAdd(&cntN[batch[v]], 1.f);
}

__global__ void gather_cg(const float* __restrict__ cemb, const int* __restrict__ batch,
                          float* __restrict__ cg) {
  int idx = blockIdx.x * blockDim.x + threadIdx.x;
  if (idx >= NTOT * EMB) return;
  int v = idx >> 6, c = idx & 63;
  int g = (v < NND) ? batch[v] : (v - NND);
  cg[idx] = cemb[(size_t)g * EMB + c];
}

__global__ void ea_fill(const float* __restrict__ eemb, float* __restrict__ ea) {
  int idx = blockIdx.x * blockDim.x + threadIdx.x;
  if (idx >= EVN * EMB) return;
  int e = idx >> 6, c = idx & 63;
  ea[idx] = (e < NE) ? eemb[idx] : (c == 0 ? 1.f : 0.f);
}

__global__ void loop_accum(const float* __restrict__ eemb, const int* __restrict__ ei,
                           const int* __restrict__ batch,
                           float* __restrict__ ea_loop, float* __restrict__ deg) {
  int idx = blockIdx.x * blockDim.x + threadIdx.x;
  if (idx >= EVN * EMB) return;
  int e = idx >> 6, c = idx & 63;
  int s, d; edge_sd(e, ei, batch, s, d); (void)s;
  if (e < NE)        atomicAdd(&ea_loop[(size_t)d * EMB + c], eemb[idx]);
  else if (c == 0)   atomicAdd(&ea_loop[(size_t)d * EMB], 1.f);
  if (c == 0)        atomicAdd(&deg[d], 1.f);
}

__global__ void loop_div(float* __restrict__ ea_loop, const float* __restrict__ deg) {
  int idx = blockIdx.x * blockDim.x + threadIdx.x;
  if (idx >= NTOT * EMB) return;
  ea_loop[idx] /= fmaxf(deg[idx >> 6], 1.f);
}

// pyg graph-LayerNorm (affine=False): scalar mean/var per graph over nodes+channels
__global__ void ln_accum(const float* __restrict__ src, int lds,
                         const int* __restrict__ batch,
                         float* __restrict__ gsum, float* __restrict__ gss) {
  int v = blockIdx.x * blockDim.x + threadIdx.x;
  if (v >= NTOT) return;
  int g = (v < NND) ? batch[v] : (v - NND);
  const float* row = src + (size_t)v * lds;
  float s = 0.f, ss = 0.f;
  for (int c = 0; c < EMB; ++c) { float t = row[c]; s += t; ss += t * t; }
  atomicAdd(&gsum[g], s);
  atomicAdd(&gss[g], ss);
}

__global__ void ln_norm(const float* __restrict__ src, int lds,
                        float* __restrict__ dst, int ldd,
                        const int* __restrict__ batch,
                        const float* __restrict__ gsum, const float* __restrict__ gss,
                        const float* __restrict__ cntA) {
  int idx = blockIdx.x * blockDim.x + threadIdx.x;
  if (idx >= NTOT * EMB) return;
  int v = idx >> 6, c = idx & 63;
  int g = (v < NND) ? batch[v] : (v - NND);
  float inv  = 1.f / (cntA[g] * (float)EMB);
  float mean = gsum[g] * inv;
  float var  = gss[g] * inv - mean * mean;
  dst[(size_t)v * ldd + c] = (src[(size_t)v * lds + c] - mean) * rsqrtf(var + EPS_LN);
}

// GENConv: msg = relu(o_n[src] + ea) + eps ; scatter-add to dst
__global__ void gen_msg(const float* __restrict__ xin, const float* __restrict__ ea,
                        const int* __restrict__ ei, const int* __restrict__ batch,
                        float* __restrict__ agg) {
  int idx = blockIdx.x * blockDim.x + threadIdx.x;
  if (idx >= EAUG * EMB) return;
  int e = idx >> 6, c = idx & 63;
  int s, d; edge_sd(e, ei, batch, s, d);
  float m = xin[(size_t)s * HC + c] + ea[idx];
  m = fmaxf(m, 0.f) + EPS_GEN;
  atomicAdd(&agg[(size_t)d * EMB + c], m);
}

__global__ void agg_add_on(const float* __restrict__ xin, float* __restrict__ agg) {
  int idx = blockIdx.x * blockDim.x + threadIdx.x;
  if (idx >= NTOT * EMB) return;
  int v = idx >> 6, c = idx & 63;
  agg[idx] += xin[(size_t)v * HC + c];
}

__global__ void attn_score(const float* __restrict__ qb, const float* __restrict__ kb,
                           const float* __restrict__ eh,
                           const int* __restrict__ ei, const int* __restrict__ batch,
                           float* __restrict__ alp, float* __restrict__ amax) {
  int idx = blockIdx.x * blockDim.x + threadIdx.x;
  if (idx >= EAUG * 2) return;
  int e = idx >> 1, h = idx & 1;
  int s, d; edge_sd(e, ei, batch, s, d);
  const float* qp = qb + (size_t)d * HC + h * EMB;
  const float* kp = kb + (size_t)s * HC + h * EMB;
  const float* ep = eh + (size_t)e * HC + h * EMB;
  float sum = 0.f;
  for (int c = 0; c < EMB; ++c) sum += qp[c] * (kp[c] + ep[c]);
  sum *= 0.125f;                    // 1/sqrt(EMB)
  alp[idx] = sum;
  atomicMaxF(&amax[d * 2 + h], sum);
}

__global__ void attn_exp(const float* __restrict__ alp, const float* __restrict__ amax,
                         const int* __restrict__ ei, const int* __restrict__ batch,
                         float* __restrict__ aex, float* __restrict__ den) {
  int idx = blockIdx.x * blockDim.x + threadIdx.x;
  if (idx >= EAUG * 2) return;
  int e = idx >> 1, h = idx & 1;
  int s, d; edge_sd(e, ei, batch, s, d); (void)s;
  float a = __expf(alp[idx] - amax[d * 2 + h]);
  aex[idx] = a;
  atomicAdd(&den[d * 2 + h], a);
}

__global__ void attn_aggregate(const float* __restrict__ vb, const float* __restrict__ eh,
                               const float* __restrict__ aex, const float* __restrict__ den,
                               const int* __restrict__ ei, const int* __restrict__ batch,
                               float* __restrict__ tout) {
  int idx = blockIdx.x * blockDim.x + threadIdx.x;
  if (idx >= EAUG * HC) return;
  int e = idx >> 7, rc = idx & 127;
  int h = rc >> 6;
  int s, d; edge_sd(e, ei, batch, s, d);
  float w = aex[e * 2 + h] / (den[d * 2 + h] + 1e-16f);
  atomicAdd(&tout[(size_t)d * HC + rc],
            (vb[(size_t)s * HC + rc] + eh[(size_t)e * HC + rc]) * w);
}

__global__ void film_update(const float* __restrict__ lbuf, const float* __restrict__ csb,
                            float* __restrict__ o) {
  int idx = blockIdx.x * blockDim.x + threadIdx.x;
  if (idx >= NTOT * EMB) return;
  int v = idx >> 6, c = idx & 63;
  o[idx] += lbuf[idx] * csb[(size_t)v * HC + c] + csb[(size_t)v * HC + EMB + c];
}

__global__ void pool_accum(const float* __restrict__ o, const int* __restrict__ batch,
                           float* __restrict__ pool) {
  int idx = blockIdx.x * blockDim.x + threadIdx.x;
  if (idx >= NND * EMB) return;
  int v = idx >> 6;
  atomicAdd(&pool[(size_t)batch[v] * EMB + (idx & 63)], o[idx]);
}

__global__ void write_out2(const float* __restrict__ pool, const float* __restrict__ cntN,
                           const float* __restrict__ o, float* __restrict__ out2) {
  int idx = blockIdx.x * blockDim.x + threadIdx.x;
  if (idx >= NG * EMB) return;
  int g = idx >> 6, c = idx & 63;
  out2[(size_t)g * HC + c]       = pool[idx] / fmaxf(cntN[g], 1.f);
  out2[(size_t)g * HC + EMB + c] = o[(size_t)(NND + g) * EMB + c];
}

// ---------------------------------------------------------------------------
extern "C" void kernel_launch(void* const* d_in, const int* in_sizes, int n_in,
                              void* d_out, int out_size, void* d_ws, size_t ws_size,
                              hipStream_t stream)
{
  (void)in_sizes; (void)n_in; (void)out_size; (void)ws_size;

  const float* x     = (const float*)d_in[0];
  const float* eatt  = (const float*)d_in[1];
  const float* cond  = (const float*)d_in[2];
  const int*   ei    = (const int*)d_in[3];
  const int*   batch = (const int*)d_in[4];
  // d_in[5] = num_graphs (compile-time constant NG)
  const float* x2h_w0=(const float*)d_in[6],  *x2h_b0=(const float*)d_in[7];
  const float* x2h_w1=(const float*)d_in[8],  *x2h_b1=(const float*)d_in[9];
  const float* x2h_w2=(const float*)d_in[10], *x2h_b2=(const float*)d_in[11];
  const float* e2h_w0=(const float*)d_in[12], *e2h_b0=(const float*)d_in[13];
  const float* e2h_w1=(const float*)d_in[14], *e2h_b1=(const float*)d_in[15];
  const float* e2h_w2=(const float*)d_in[16], *e2h_b2=(const float*)d_in[17];
  const float* c2h_w0=(const float*)d_in[18], *c2h_b0=(const float*)d_in[19];
  const float* c2h_w1=(const float*)d_in[20], *c2h_b1=(const float*)d_in[21];
  const float* c2h_w2=(const float*)d_in[22], *c2h_b2=(const float*)d_in[23];
  const float* gen_w =(const float*)d_in[24], *gen_b =(const float*)d_in[25];
  const float* tq_w  =(const float*)d_in[26], *tq_b  =(const float*)d_in[27];
  const float* tk_w  =(const float*)d_in[28], *tk_b  =(const float*)d_in[29];
  const float* tv_w  =(const float*)d_in[30], *tv_b  =(const float*)d_in[31];
  const float* ts_w  =(const float*)d_in[32], *ts_b  =(const float*)d_in[33];
  const float* te_w  =(const float*)d_in[34];
  const float* lin_w =(const float*)d_in[35], *lin_b =(const float*)d_in[36];
  const float* ff_w0 =(const float*)d_in[37], *ff_b0 =(const float*)d_in[38];
  const float* ff_w1 =(const float*)d_in[39], *ff_b1 =(const float*)d_in[40];
  const float* cs_w  =(const float*)d_in[41], *cs_b  =(const float*)d_in[42];

  // ---- workspace bump allocator --------------------------------------------
  uint8_t* wp = (uint8_t*)d_ws;
  auto alloc = [&](size_t nfloats) -> float* {
    float* r = (float*)wp;
    wp += ((nfloats * sizeof(float)) + 255) & ~(size_t)255;
    return r;
  };
  float* o    = alloc((size_t)NTOT * EMB);   // node+vnode state
  float* xin  = alloc((size_t)NTOT * HC);    // [o_n | gen out]
  float* agg  = alloc((size_t)NTOT * EMB);
  float* qb   = alloc((size_t)NTOT * HC);
  float* kb   = alloc((size_t)NTOT * HC);
  float* vb   = alloc((size_t)NTOT * HC);
  float* tout = alloc((size_t)NTOT * HC);
  float* csb  = alloc((size_t)NTOT * HC);
  float* lbuf = alloc((size_t)NTOT * EMB);
  float* hbuf = alloc((size_t)NTOT * EMB);
  float* ffh  = alloc((size_t)NTOT * 256);
  float* cemb = alloc((size_t)NG * EMB);
  float* cg   = alloc((size_t)NTOT * EMB);
  float* eemb = alloc((size_t)NE * EMB);
  float* ea   = alloc((size_t)EAUG * EMB);
  float* eh   = alloc((size_t)EAUG * HC);
  float* alp  = alloc((size_t)EAUG * 2);
  float* aex  = alloc((size_t)EAUG * 2);
  float* amax = alloc((size_t)NTOT * 2);
  float* den  = alloc((size_t)NTOT * 2);
  float* gsum = alloc(NG);
  float* gss  = alloc(NG);
  float* cntA = alloc(NG);
  float* cntN = alloc(NG);
  float* deg  = alloc(NTOT);
  float* pool = alloc((size_t)NG * EMB);
  float* t1   = eh;                          // reuse eh region for MLP temps
  float* t2   = eh + (size_t)NE * EMB;

  const dim3 blk(256, 1, 1);
  auto grid1 = [&](int n) { return dim3(CDIV(n, 256), 1, 1); };

  // ===== input MLPs (WMMA) ===================================================
  launch_gemm< 32, 1, false>(x,    x2h_w0, EMB, x2h_b0, t1,   EMB, NND, EMB, stream);
  launch_gemm< 64, 1, false>(t1,   x2h_w1, EMB, x2h_b1, t2,   EMB, NND, EMB, stream);
  launch_gemm< 64, 0, false>(t2,   x2h_w2, EMB, x2h_b2, o,    EMB, NND, EMB, stream);
  launch_gemm< 16, 1, false>(eatt, e2h_w0, EMB, e2h_b0, t1,   EMB, NE,  EMB, stream);
  launch_gemm< 64, 1, false>(t1,   e2h_w1, EMB, e2h_b1, t2,   EMB, NE,  EMB, stream);
  launch_gemm< 64, 0, false>(t2,   e2h_w2, EMB, e2h_b2, eemb, EMB, NE,  EMB, stream);
  launch_gemm< 32, 1, false>(cond, c2h_w0, EMB, c2h_b0, t1,   EMB, NG,  EMB, stream);
  launch_gemm< 64, 1, false>(t1,   c2h_w1, EMB, c2h_b1, t2,   EMB, NG,  EMB, stream);
  launch_gemm< 64, 0, false>(t2,   c2h_w2, EMB, c2h_b2, o + (size_t)NND * EMB, EMB, NG, EMB, stream);
  hipMemcpyAsync(cemb, o + (size_t)NND * EMB, (size_t)NG * EMB * sizeof(float),
                 hipMemcpyDeviceToDevice, stream);

  // ===== layer-invariant prep ===============================================
  fill_f32<<<grid1(NG), blk, 0, stream>>>(cntA, 1.f, NG);     // +1 for virtual node
  hipMemsetAsync(cntN, 0, NG * sizeof(float), stream);
  count_nodes<<<grid1(NND), blk, 0, stream>>>(batch, cntA, cntN);
  gather_cg<<<grid1(NTOT * EMB), blk, 0, stream>>>(cemb, batch, cg);

  // augmented edge attrs: rows [0,EVN) = [e | one-hot], tail = self-loop mean
  ea_fill<<<grid1(EVN * EMB), blk, 0, stream>>>(eemb, ea);
  hipMemsetAsync(ea + (size_t)EVN * EMB, 0, (size_t)NTOT * EMB * sizeof(float), stream);
  hipMemsetAsync(deg, 0, NTOT * sizeof(float), stream);
  loop_accum<<<grid1(EVN * EMB), blk, 0, stream>>>(eemb, ei, batch, ea + (size_t)EVN * EMB, deg);
  loop_div<<<grid1(NTOT * EMB), blk, 0, stream>>>(ea + (size_t)EVN * EMB, deg);

  // ===== layers ==============================================================
  for (int i = 0; i < 3; ++i) {
    const float* gw  = gen_w + (size_t)i * EMB * EMB;  const float* gb2 = gen_b + (size_t)i * EMB;
    const float* qw  = tq_w  + (size_t)i * HC  * HC;   const float* qb2 = tq_b  + (size_t)i * HC;
    const float* kw  = tk_w  + (size_t)i * HC  * HC;   const float* kb2 = tk_b  + (size_t)i * HC;
    const float* vw  = tv_w  + (size_t)i * HC  * HC;   const float* vb2 = tv_b  + (size_t)i * HC;
    const float* sw  = ts_w  + (size_t)i * HC  * HC;   const float* sb2 = ts_b  + (size_t)i * HC;
    const float* tew = te_w  + (size_t)i * EMB * HC;
    const float* lw  = lin_w + (size_t)i * HC  * EMB;  const float* lb2 = lin_b + (size_t)i * EMB;
    const float* f0w = ff_w0 + (size_t)i * EMB * 256;  const float* f0b = ff_b0 + (size_t)i * 256;
    const float* f1w = ff_w1 + (size_t)i * 256 * EMB;  const float* f1b = ff_b1 + (size_t)i * EMB;
    const float* cw  = cs_w  + (size_t)i * EMB * HC;   const float* cb2 = cs_b  + (size_t)i * HC;

    // pre-LN -> xin[:, :64]
    hipMemsetAsync(gsum, 0, NG * sizeof(float), stream);
    hipMemsetAsync(gss,  0, NG * sizeof(float), stream);
    ln_accum<<<grid1(NTOT), blk, 0, stream>>>(o, EMB, batch, gsum, gss);
    ln_norm<<<grid1(NTOT * EMB), blk, 0, stream>>>(o, EMB, xin, HC, batch, gsum, gss, cntA);

    // GENConv -> xin[:, 64:]
    hipMemsetAsync(agg, 0, (size_t)NTOT * EMB * sizeof(float), stream);
    gen_msg<<<grid1(EAUG * EMB), blk, 0, stream>>>(xin, ea, ei, batch, agg);
    agg_add_on<<<grid1(NTOT * EMB), blk, 0, stream>>>(xin, agg);
    launch_gemm< 64, 0, false>(agg, gw, EMB, gb2, xin + EMB, HC, NTOT, EMB, stream);

    // TransformerConv projections (WMMA)
    launch_gemm<128, 0, false>(xin, qw, HC, qb2, qb, HC, NTOT, HC, stream);
    launch_gemm<128, 0, false>(xin, kw, HC, kb2, kb, HC, NTOT, HC, stream);
    launch_gemm<128, 0, false>(xin, vw, HC, vb2, vb, HC, NTOT, HC, stream);
    launch_gemm< 64, 0, false>(ea, tew, HC, nullptr, eh, HC, EAUG, HC, stream);

    // segment softmax attention
    fill_f32<<<grid1(NTOT * 2), blk, 0, stream>>>(amax, -3e38f, NTOT * 2);
    hipMemsetAsync(den, 0, (size_t)NTOT * 2 * sizeof(float), stream);
    attn_score<<<grid1(EAUG * 2), blk, 0, stream>>>(qb, kb, eh, ei, batch, alp, amax);
    attn_exp<<<grid1(EAUG * 2), blk, 0, stream>>>(alp, amax, ei, batch, aex, den);
    hipMemsetAsync(tout, 0, (size_t)NTOT * HC * sizeof(float), stream);
    attn_aggregate<<<grid1(EAUG * HC), blk, 0, stream>>>(vb, eh, aex, den, ei, batch, tout);
    launch_gemm<128, 0, true >(xin, sw, HC, sb2, tout, HC, NTOT, HC, stream);   // += root skip
    launch_gemm<128, 0, false>(tout, lw, EMB, lb2, lbuf, EMB, NTOT, EMB, stream);

    // FiLM conditioning + residual
    launch_gemm< 64, 0, false>(cg, cw, HC, cb2, csb, HC, NTOT, HC, stream);
    film_update<<<grid1(NTOT * EMB), blk, 0, stream>>>(lbuf, csb, o);

    // FF block: o += W1 @ leaky(W0 @ LN(o))
    hipMemsetAsync(gsum, 0, NG * sizeof(float), stream);
    hipMemsetAsync(gss,  0, NG * sizeof(float), stream);
    ln_accum<<<grid1(NTOT), blk, 0, stream>>>(o, EMB, batch, gsum, gss);
    ln_norm<<<grid1(NTOT * EMB), blk, 0, stream>>>(o, EMB, hbuf, EMB, batch, gsum, gss, cntA);
    launch_gemm< 64, 1, false>(hbuf, f0w, 256, f0b, ffh, 256, NTOT, 256, stream);
    launch_gemm<256, 0, true >(ffh, f1w, EMB, f1b, o, EMB, NTOT, EMB, stream);
  }

  // ===== outputs =============================================================
  hipMemsetAsync(pool, 0, (size_t)NG * EMB * sizeof(float), stream);
  pool_accum<<<grid1(NND * EMB), blk, 0, stream>>>(o, batch, pool);
  hipMemcpyAsync(d_out, o, (size_t)NND * EMB * sizeof(float),
                 hipMemcpyDeviceToDevice, stream);
  write_out2<<<grid1(NG * EMB), blk, 0, stream>>>(pool, cntN, o,
                                                  (float*)d_out + (size_t)NND * EMB);
}
